// MultiHeadAttention_17497696764227
// MI455X (gfx1250) — matrix operand order repaired and checked by reading, not compiled, not run on hardware
//
#include <hip/hip_runtime.h>

// ---------------- types ----------------
typedef __bf16  v16bf __attribute__((ext_vector_type(16)));
typedef __bf16  v8bf  __attribute__((ext_vector_type(8)));
typedef float   v8f   __attribute__((ext_vector_type(8)));

#define BB 16
#define CC 256
#define HH 64
#define WW 64
#define HWSZ 4096          // H*W
#define NTOK 65536         // B*H*W
#define D3 768             // 3*C
#define NHEAD 8
#define HDIM 32

// ---------------- WMMA helpers (layouts per CDNA5 ISA §7.12.2) ----
__device__ __forceinline__ v8f wmma_bf16(v16bf a, v16bf b, v8f c) {
    // (neg_a, A, neg_b, B, c_mod, C, reuse_a, reuse_b)
    return __builtin_amdgcn_wmma_f32_16x16x32_bf16(false, a, false, b, (short)0, c, false, false);
}

// A-matrix 16x32 bf16, row-major source with leading dim ld (elements).
// lane<16: row=lane, K={0..7,16..23}; lane>=16: row=lane-16, K={8..15,24..31}
__device__ __forceinline__ v16bf load_a_frag(const __bf16* base, int ld, int lane) {
    int row  = lane & 15;
    int koff = (lane < 16) ? 0 : 8;
    const __bf16* p = base + (size_t)row * ld + koff;
    union { v16bf v; v8bf h[2]; } u;
    u.h[0] = *(const v8bf*)(p);
    u.h[1] = *(const v8bf*)(p + 16);
    return u.v;
}

// B-matrix 32x16 bf16 where source rows are the N dimension (i.e. B(k,n) = src[n][k]).
// lane<16: col=lane, K={0..7,16..23}; lane>=16: col=lane-16, K={8..15,24..31}
__device__ __forceinline__ v16bf load_bt_frag(const __bf16* base, int ld, int lane) {
    int col  = lane & 15;
    int koff = (lane < 16) ? 0 : 8;
    const __bf16* p = base + (size_t)col * ld + koff;
    union { v16bf v; v8bf h[2]; } u;
    u.h[0] = *(const v8bf*)(p);
    u.h[1] = *(const v8bf*)(p + 16);
    return u.v;
}

// ---------------- kernel 0: weights fp32 -> bf16 ----------------
__global__ void cvt_weights(const float* __restrict__ wq, const float* __restrict__ wp,
                            __bf16* __restrict__ wqb, __bf16* __restrict__ wpb) {
    int i = blockIdx.x * blockDim.x + threadIdx.x;
    int stride = gridDim.x * blockDim.x;
    for (int j = i; j < D3 * CC; j += stride) wqb[j] = (__bf16)wq[j];
    for (int j = i; j < CC * CC; j += stride) wpb[j] = (__bf16)wp[j];
}

// ---------------- kernel 1: LayerNorm + channel transpose, bf16 out ----------
// block = 256 threads, handles 32 tokens of one (b,h) row (half a row).
__global__ void ln_kernel(const float* __restrict__ x, const float* __restrict__ gamma,
                          const float* __restrict__ beta, __bf16* __restrict__ xn) {
    __shared__ float tile[32][257];
    __shared__ float s_mu[32], s_rs[32];
    int blk  = blockIdx.x;         // 0..2047
    int bh   = blk >> 1;           // 0..1023
    int half = blk & 1;
    int b = bh >> 6, h = bh & 63;
    int tid = threadIdx.x;

    const float* xb = x + (size_t)b * CC * HWSZ + h * WW + half * 32;
    for (int i = 0; i < 32; ++i) {               // 32 tokens x 256 ch, coalesced 128B runs
        int idx = i * 256 + tid;
        int c   = idx >> 5;
        int w0  = idx & 31;
        tile[w0][c] = xb[(size_t)c * HWSZ + w0];
    }
    __syncthreads();
    if (tid < 32) {
        float s = 0.f, s2 = 0.f;
        for (int c = 0; c < CC; ++c) { float v = tile[tid][c]; s += v; s2 += v * v; }
        float mu  = s * (1.f / CC);
        float var = s2 * (1.f / CC) - mu * mu;
        s_mu[tid] = mu;
        s_rs[tid] = rsqrtf(var + 1e-5f);
    }
    __syncthreads();
    float g  = gamma[tid];
    float be = beta[tid];
    int t0 = bh * 64 + half * 32;
    for (int i = 0; i < 32; ++i) {
        float v = (tile[i][tid] - s_mu[i]) * s_rs[i] * g + be;
        xn[(size_t)(t0 + i) * CC + tid] = (__bf16)v;
    }
}

// ---------------- kernel 2: QKV GEMM (N x 256) * (768 x 256)^T -> bf16 -------
// block 256 thr = 8 waves in 2x4; block tile 64 tokens x 128 cols; wave tile 32x32.
// C-fragments staged through LDS so the 96 MB qkv stream is written in 256B runs.
__global__ void qkv_gemm(const __bf16* __restrict__ A, const __bf16* __restrict__ Wq,
                         __bf16* __restrict__ Q) {
    __shared__ __bf16 otile[64][136];
    int tid  = threadIdx.x;
    int lane = tid & 31;
    int wid  = tid >> 5;
    int wm = wid >> 2, wn = wid & 3;
    int t0b = blockIdx.y * 64;
    int n0b = blockIdx.x * 128;
    int t0 = t0b + wm * 32;
    int n0 = n0b + wn * 32;

    v8f acc[2][2];
    for (int i = 0; i < 2; ++i)
        for (int j = 0; j < 2; ++j)
            for (int v = 0; v < 8; ++v) acc[i][j][v] = 0.f;

    for (int kk = 0; kk < CC; kk += 32) {
        v16bf a0 = load_a_frag (A  + (size_t)t0 * CC        + kk, CC, lane);
        v16bf a1 = load_a_frag (A  + (size_t)(t0 + 16) * CC + kk, CC, lane);
        v16bf b0 = load_bt_frag(Wq + (size_t)n0 * CC        + kk, CC, lane);
        v16bf b1 = load_bt_frag(Wq + (size_t)(n0 + 16) * CC + kk, CC, lane);
        acc[0][0] = wmma_bf16(a0, b0, acc[0][0]);
        acc[0][1] = wmma_bf16(a0, b1, acc[0][1]);
        acc[1][0] = wmma_bf16(a1, b0, acc[1][0]);
        acc[1][1] = wmma_bf16(a1, b1, acc[1][1]);
    }
    int r = 8 * (lane >> 4), cN = lane & 15;
    for (int mi = 0; mi < 2; ++mi)
        for (int ni = 0; ni < 2; ++ni) {
            int row_l = wm * 32 + mi * 16 + r;
            int col_l = wn * 32 + ni * 16 + cN;
            for (int v = 0; v < 8; ++v)
                otile[row_l + v][col_l] = (__bf16)acc[mi][ni][v];
        }
    __syncthreads();
    for (int idx = tid; idx < 64 * 128; idx += 256) {
        int row = idx >> 7, col = idx & 127;
        Q[(size_t)(t0b + row) * D3 + n0b + col] = otile[row][col];
    }
}

// ---------------- kernel 3: attention, 2 waves per (bh, head) ---------------
__global__ void __launch_bounds__(64)
attn_kernel(const __bf16* __restrict__ qkv, __bf16* __restrict__ ao) {
    __shared__ float  S[64][65];
    __shared__ __bf16 P[64][80];
    __shared__ __bf16 VT[32][80];
    int tid  = threadIdx.x;        // 0..63
    int lane = tid & 31;
    int wv   = tid >> 5;           // wave id 0..1
    int prob = blockIdx.x;
    int bh   = prob >> 3;
    int head = prob & 7;
    size_t t0 = (size_t)bh * 64;
    const __bf16* Qb = qkv + t0 * D3 + head * HDIM;
    const __bf16* Kb = Qb + CC;
    const __bf16* Vb = Qb + 2 * CC;
    const float scale = 0.17677669529663687f;   // 1/sqrt(32)
    int r = 8 * (lane >> 4), cN = lane & 15;

    // S = Q K^T  (head_dim 32 == one WMMA K-step); wave wv owns ni = wv*2+{0,1}
    v16bf aQ[4];
    for (int mi = 0; mi < 4; ++mi)
        aQ[mi] = load_a_frag(Qb + (size_t)(mi * 16) * D3, D3, lane);
    for (int j = 0; j < 2; ++j) {
        int ni = wv * 2 + j;
        v16bf bK = load_bt_frag(Kb + (size_t)(ni * 16) * D3, D3, lane);
        for (int mi = 0; mi < 4; ++mi) {
            v8f acc; for (int v = 0; v < 8; ++v) acc[v] = 0.f;
            acc = wmma_bf16(aQ[mi], bK, acc);
            for (int v = 0; v < 8; ++v)
                S[mi * 16 + r + v][ni * 16 + cN] = acc[v] * scale;
        }
    }
    // stage V transposed into LDS (coalesced global reads)
    for (int i = 0; i < 32; ++i) {
        int idx = i * 64 + tid;
        int key = idx >> 5, d = idx & 31;
        VT[d][key] = Vb[(size_t)key * D3 + d];
    }
    __syncthreads();
    // softmax: one row per thread
    {
        int rr = tid;
        float mx = -3.4e38f;
        for (int k = 0; k < 64; ++k) mx = fmaxf(mx, S[rr][k]);
        float sum = 0.f;
        for (int k = 0; k < 64; ++k) { float e = __expf(S[rr][k] - mx); S[rr][k] = e; sum += e; }
        float inv = 1.f / sum;
        for (int k = 0; k < 64; ++k) P[rr][k] = (__bf16)(S[rr][k] * inv);
    }
    __syncthreads();
    // O = P V  (K = 64 -> two WMMA steps; N = 32 -> two tiles); wave owns mi = wv*2+{0,1}
    // O buffer aliases S (dead after softmax): 64 x 40 bf16 = 5 KB << 16.6 KB.
    __bf16* Obuf = (__bf16*)&S[0][0];
    for (int jm = 0; jm < 2; ++jm) {
        int mi = wv * 2 + jm;
        for (int ni = 0; ni < 2; ++ni) {
            v8f acc; for (int v = 0; v < 8; ++v) acc[v] = 0.f;
            for (int k0 = 0; k0 < 64; k0 += 32) {
                v16bf aP = load_a_frag (&P [mi * 16][0] + k0, 80, lane);
                v16bf bV = load_bt_frag(&VT[ni * 16][0] + k0, 80, lane);
                acc = wmma_bf16(aP, bV, acc);
            }
            for (int v = 0; v < 8; ++v)
                Obuf[(mi * 16 + r + v) * 40 + ni * 16 + cN] = (__bf16)acc[v];
        }
    }
    __syncthreads();
    // coalesced 64B-run write-out of the 32 MB attention-out stream
    for (int i = 0; i < 32; ++i) {
        int idx = i * 64 + tid;
        int row = idx >> 5, col = idx & 31;
        ao[(t0 + row) * CC + head * HDIM + col] = Obuf[row * 40 + col];
    }
}

// ---------------- kernel 4: proj GEMM + bias + transpose to (B,C,H,W) -------
__global__ void proj_gemm(const __bf16* __restrict__ A, const __bf16* __restrict__ Wp,
                          const float* __restrict__ bias, float* __restrict__ out) {
    __shared__ float otile[128][65];     // [col_local][token_local]
    int tid  = threadIdx.x;
    int lane = tid & 31;
    int wid  = tid >> 5;
    int wm = wid >> 2, wn = wid & 3;
    int t0 = blockIdx.y * 64;            // one full (b,h) row of 64 tokens
    int tw = t0 + wm * 32;
    int n0 = blockIdx.x * 128;
    int nw = n0 + wn * 32;

    v8f acc[2][2];
    for (int i = 0; i < 2; ++i)
        for (int j = 0; j < 2; ++j)
            for (int v = 0; v < 8; ++v) acc[i][j][v] = 0.f;

    for (int kk = 0; kk < CC; kk += 32) {
        v16bf a0 = load_a_frag (A  + (size_t)tw * CC        + kk, CC, lane);
        v16bf a1 = load_a_frag (A  + (size_t)(tw + 16) * CC + kk, CC, lane);
        v16bf b0 = load_bt_frag(Wp + (size_t)nw * CC        + kk, CC, lane);
        v16bf b1 = load_bt_frag(Wp + (size_t)(nw + 16) * CC + kk, CC, lane);
        acc[0][0] = wmma_bf16(a0, b0, acc[0][0]);
        acc[0][1] = wmma_bf16(a0, b1, acc[0][1]);
        acc[1][0] = wmma_bf16(a1, b0, acc[1][0]);
        acc[1][1] = wmma_bf16(a1, b1, acc[1][1]);
    }
    int r = 8 * (lane >> 4), cN = lane & 15;
    for (int mi = 0; mi < 2; ++mi)
        for (int ni = 0; ni < 2; ++ni) {
            int col_l = wn * 32 + ni * 16 + cN;
            int row_l = wm * 32 + mi * 16 + r;
            float bb = bias[n0 + col_l];
            for (int v = 0; v < 8; ++v)
                otile[col_l][row_l + v] = acc[mi][ni][v] + bb;
        }
    __syncthreads();
    int b   = t0 >> 12;       // t0 / 4096
    int hw0 = t0 & 4095;
    for (int idx = tid; idx < 128 * 64; idx += 256) {
        int dl = idx >> 6, w0 = idx & 63;
        out[((size_t)(b * CC + n0 + dl)) * HWSZ + hw0 + w0] = otile[dl][w0];
    }
}

// ---------------- launch ----------------
extern "C" void kernel_launch(void* const* d_in, const int* in_sizes, int n_in,
                              void* d_out, int out_size, void* d_ws, size_t ws_size,
                              hipStream_t stream) {
    (void)in_sizes; (void)n_in; (void)out_size; (void)ws_size;
    const float* x      = (const float*)d_in[0];
    const float* w_qkv  = (const float*)d_in[1];
    const float* w_proj = (const float*)d_in[2];
    const float* b_proj = (const float*)d_in[3];
    const float* gamma  = (const float*)d_in[4];
    const float* beta   = (const float*)d_in[5];
    float* out = (float*)d_out;

    char* ws = (char*)d_ws;
    __bf16* xn  = (__bf16*)(ws);                                    //  32 MiB
    __bf16* qkv = (__bf16*)(ws + ((size_t)32  << 20));              //  96 MiB
    __bf16* ao  = (__bf16*)(ws + ((size_t)128 << 20));              //  32 MiB
    __bf16* wqb = (__bf16*)(ws + ((size_t)160 << 20));              // 384 KiB
    __bf16* wpb = (__bf16*)(ws + ((size_t)160 << 20) + (1u << 20)); // 128 KiB

    cvt_weights<<<dim3(256), dim3(256), 0, stream>>>(w_qkv, w_proj, wqb, wpb);
    ln_kernel  <<<dim3(2048), dim3(256), 0, stream>>>(x, gamma, beta, xn);
    qkv_gemm   <<<dim3(D3 / 128, NTOK / 64), dim3(256), 0, stream>>>(xn, wqb, qkv);
    attn_kernel<<<dim3(1024 * NHEAD), dim3(64), 0, stream>>>(qkv, ao);
    proj_gemm  <<<dim3(CC / 128, NTOK / 64), dim3(256), 0, stream>>>(ao, wpb, b_proj, out);
}
